// EdgeGuidedLocalSSI_35648228557009
// MI455X (gfx1250) — compile-verified, matrix operand order checked by reading, not run on hardware
//
#include <hip/hip_runtime.h>
#include <hip/hip_bf16.h>

// ---------------- problem constants (match reference setup) ----------------
#define B_      32
#define H_      480
#define W_      640
#define OH_     34
#define OW_     45
#define NP_     (OH_*OW_)        // 1530
#define K_      153              // max(int(0.1*1530),10)
#define PS_     25               // kernel size: int(0.05*480)=24 -> odd -> 25
#define PH_     12
#define PATCH_  (PS_*PS_)        // 625
#define NTOT_   (H_*W_)          // 307200
#define CHUNK_  2048
#define NCHUNK_ 150              // 150*2048 == 307200
#define RESHAPE_ 14
#define EPS_    1e-5f

typedef float    v2f  __attribute__((ext_vector_type(2)));
typedef float    v8f  __attribute__((ext_vector_type(8)));
typedef _Float16 v16h __attribute__((ext_vector_type(16)));

// ---------------- float <-> ascending-sortable key ----------------
__device__ __forceinline__ unsigned fkey(float x) {
  unsigned u = __float_as_uint(x);
  return (u & 0x80000000u) ? ~u : (u | 0x80000000u);
}

// ---------------- WMMA-based wave (32-lane) reductions ----------------
// A-layout (32-bit A 16x4): lane m (0..15) -> A[m][K0] in vgpr0, lanes 16..31 -> A[m][K2].
// With B = ones: D[i][j] = v(lane i) + v(lane 16+i) for all j.
// D-layout: lane j<16 holds rows 0..7 in c[0..7]; lanes 16..31 hold rows 8..15.
// => hsum(c) + shfl_xor(16) == full wave sum (exact f32).
__device__ __forceinline__ float wave_sum_f32(float v) {
#if __has_builtin(__builtin_amdgcn_wmma_f32_16x16x4_f32)
  v2f a; a[0] = v;    a[1] = 0.0f;
  v2f b; b[0] = 1.0f; b[1] = 1.0f;
  v8f c = {};
  c = __builtin_amdgcn_wmma_f32_16x16x4_f32(false, a, false, b, (short)0, c, false, false);
  float s = ((c[0] + c[1]) + (c[2] + c[3])) + ((c[4] + c[5]) + (c[6] + c[7]));
  s += __shfl_xor(s, 16, 32);
  return s;
#else
  for (int off = 16; off; off >>= 1) v += __shfl_xor(v, off, 32);
  return v;
#endif
}

// Exact for small non-negative integers (f16 inputs hold ints<=2048 exactly,
// accumulation is f32). Uses the codegen-confirmed 16x16x32 f16 WMMA.
// 16-bit A layout: lane m<16: half[0] = A[m][K=0]; lane 16+m: half[0] = A[m][K=8].
__device__ __forceinline__ float wave_sum_int_f16(float v) {
  v16h a = {};
  a[0] = (_Float16)v;
  v16h b;
#pragma unroll
  for (int i = 0; i < 16; ++i) b[i] = (_Float16)1.0f;
  v8f c = {};
  c = __builtin_amdgcn_wmma_f32_16x16x32_f16(false, a, false, b, (short)0, c, false, false);
  float s = ((c[0] + c[1]) + (c[2] + c[3])) + ((c[4] + c[5]) + (c[6] + c[7]));
  s += __shfl_xor(s, 16, 32);
  return s;
}

__device__ __forceinline__ float block_sum(float v, float* red) {
  int tid = threadIdx.x;
  float s = wave_sum_f32(v);
  if ((tid & 31) == 0) red[tid >> 5] = s;
  __syncthreads();
  if (tid == 0) {
    float t = 0.f;
    int nw = (int)(blockDim.x >> 5);
    for (int i = 0; i < nw; ++i) t += red[i];
    red[8] = t;
  }
  __syncthreads();
  float r = red[8];
  __syncthreads();
  return r;
}

__device__ __forceinline__ float block_count(float v, float* red) {
  int tid = threadIdx.x;
  float s = wave_sum_int_f16(v);
  if ((tid & 31) == 0) red[tid >> 5] = s;
  __syncthreads();
  if (tid == 0) {
    float t = 0.f;
    int nw = (int)(blockDim.x >> 5);
    for (int i = 0; i < nw; ++i) t += red[i];
    red[8] = t;
  }
  __syncthreads();
  float r = red[8];
  __syncthreads();
  return r;
}

// ---------------- bitonic sorts in LDS ----------------
__device__ void bitonic_f(float* s, int n, int tid, int nthr) {
  for (int k = 2; k <= n; k <<= 1) {
    for (int j = k >> 1; j > 0; j >>= 1) {
      for (int t = tid; t < n; t += nthr) {
        int ixj = t ^ j;
        if (ixj > t) {
          float a = s[t], b = s[ixj];
          if ((a > b) == ((t & k) == 0)) { s[t] = b; s[ixj] = a; }
        }
      }
      __syncthreads();
    }
  }
}

// ---------------- kernel 0: zero the global histograms ----------------
__global__ void zero_hist_kernel(unsigned* __restrict__ p, int n) {
  int i = blockIdx.x * 256 + threadIdx.x;
  if (i < n) p[i] = 0u;
}

// ---------------- fused edges + erode + border + bilinear resize ----------------
__device__ float edge_at(const float* __restrict__ img,
                         const unsigned char* __restrict__ vmk, int b, int y, int x) {
  if (y < 3 || y >= H_ - 3 || x < 3 || x >= W_ - 3) return 0.f;   // bm
  const unsigned char* vm = vmk + (size_t)b * NTOT_;
#pragma unroll
  for (int dy = -1; dy <= 1; ++dy)
#pragma unroll
    for (int dx = -1; dx <= 1; ++dx)
      if (!vm[(y + dy) * W_ + (x + dx)]) return 0.f;              // erode3(vmask)
  const float* base = img + (size_t)b * 3 * NTOT_;
  float sgx = 0.f, sgy = 0.f;
#pragma unroll
  for (int c = 0; c < 3; ++c) {
    const float* p = base + (size_t)c * NTOT_;
    float a00 = p[(y - 1) * W_ + x - 1], a01 = p[(y - 1) * W_ + x], a02 = p[(y - 1) * W_ + x + 1];
    float a10 = p[y * W_ + x - 1],                                  a12 = p[y * W_ + x + 1];
    float a20 = p[(y + 1) * W_ + x - 1], a21 = p[(y + 1) * W_ + x], a22 = p[(y + 1) * W_ + x + 1];
    float gx = (-a00 + a02 - 2.f * a10 + 2.f * a12 - a20 + a22) * 0.125f;
    float gy = (-a00 - 2.f * a01 - a02 + a20 + 2.f * a21 + a22) * 0.125f;
    sgx += gx * gx; sgy += gy * gy;
  }
  return sqrtf((sgx + sgy) * (1.f / 3.f));
}

__global__ void resize_edges_kernel(const float* __restrict__ img,
                                    const unsigned char* __restrict__ vmk,
                                    float* __restrict__ e_rs) {
  int b = blockIdx.y;
  int o = blockIdx.x * 256 + threadIdx.x;
  if (o >= NP_) return;
  int oy = o / OW_, ox = o - oy * OW_;
  float cy = fminf(fmaxf((oy + 0.5f) * ((float)H_ / OH_) - 0.5f, 0.f), (float)(H_ - 1));
  int y0 = (int)floorf(cy); int y1 = min(y0 + 1, H_ - 1); float wy = cy - (float)y0;
  float cx = fminf(fmaxf((ox + 0.5f) * ((float)W_ / OW_) - 0.5f, 0.f), (float)(W_ - 1));
  int x0 = (int)floorf(cx); int x1 = min(x0 + 1, W_ - 1); float wx = cx - (float)x0;
  float v00 = edge_at(img, vmk, b, y0, x0);
  float v01 = edge_at(img, vmk, b, y0, x1);
  float v10 = edge_at(img, vmk, b, y1, x0);
  float v11 = edge_at(img, vmk, b, y1, x1);
  e_rs[b * NP_ + o] = v00 * (1.f - wy) * (1.f - wx) + v01 * (1.f - wy) * wx
                    + v10 * wy * (1.f - wx) + v11 * wy * wx;
}

// ---------------- top-k (per batch) via 2048-key bitonic sort ----------------
__global__ void __launch_bounds__(1024)
topk_kernel(const float* __restrict__ e_rs, int* __restrict__ coords) {
  __shared__ unsigned long long key[2048];
  int b = blockIdx.x, tid = threadIdx.x;
  for (int t = tid; t < 2048; t += 1024) {
    unsigned long long kk = 0ull;
    if (t < NP_) {
      unsigned vk = fkey(e_rs[b * NP_ + t]);
      kk = ((unsigned long long)vk << 32) | (unsigned long long)(0xFFFFFFFFu - (unsigned)t);
    }
    key[t] = kk;
  }
  __syncthreads();
  for (int k = 2; k <= 2048; k <<= 1) {
    for (int j = k >> 1; j > 0; j >>= 1) {
      for (int t = tid; t < 2048; t += 1024) {
        int ixj = t ^ j;
        if (ixj > t) {
          unsigned long long a = key[t], c = key[ixj];
          if ((a > c) == ((t & k) == 0)) { key[t] = c; key[ixj] = a; }
        }
      }
      __syncthreads();
    }
  }
  if (tid < K_) {
    unsigned long long kk = key[2047 - tid];          // descending: best first
    unsigned idx = 0xFFFFFFFFu - (unsigned)(kk & 0xFFFFFFFFull);
    coords[(b * K_ + tid) * 2 + 0] = (int)(idx / OW_) * RESHAPE_;
    coords[(b * K_ + tid) * 2 + 1] = (int)(idx % OW_) * RESHAPE_;
  }
}

// ---------------- per-patch SSI error ----------------
__global__ void patch_kernel(const float* __restrict__ inp, const float* __restrict__ tgt,
                             const unsigned char* __restrict__ msk,
                             const int* __restrict__ coords,
                             float* __restrict__ ep, float* __restrict__ pvalid) {
  __shared__ float srt[1024];
  __shared__ float pin[PATCH_], ptg[PATCH_], pm[PATCH_];
  __shared__ float red[12];
  int p = blockIdx.x, b = blockIdx.y, tid = threadIdx.x;
  int cy = coords[(b * K_ + p) * 2 + 0];
  int cx = coords[(b * K_ + p) * 2 + 1];
  size_t base = (size_t)b * NTOT_;
  const float INF = __uint_as_float(0x7f800000u);

  for (int e = tid; e < PATCH_; e += 256) {
    int ey = e / PS_, ex = e - ey * PS_;
    int py = cy + ey - PH_, px = cx + ex - PH_;
    bool inb = (py >= 0 && py < H_ && px >= 0 && px < W_);
    float vi = 0.f, vt = 0.f, m = 0.f;
    if (inb) {
      size_t g = base + (size_t)py * W_ + px;
      vi = inp[g]; vt = tgt[g]; m = msk[g] ? 1.f : 0.f;
    }
    pin[e] = vi; ptg[e] = vt; pm[e] = m;
  }
  __syncthreads();

  float cl = 0.f;
  for (int e = tid; e < PATCH_; e += 256) cl += pm[e];
  float nf = block_count(cl, red);                 // exact via f16 WMMA
  int n = (int)(nf + 0.5f);
  int midx = (n > 0) ? ((n - 1) >> 1) : 0;

  for (int t = tid; t < 1024; t += 256)
    srt[t] = (t < PATCH_ && pm[t] > 0.f) ? pin[t] : INF;
  __syncthreads();
  bitonic_f(srt, 1024, tid, 256);
  float med_in = (n > 0) ? srt[midx] : 0.f;
  __syncthreads();

  for (int t = tid; t < 1024; t += 256)
    srt[t] = (t < PATCH_ && pm[t] > 0.f) ? ptg[t] : INF;
  __syncthreads();
  bitonic_f(srt, 1024, tid, 256);
  float med_tg = (n > 0) ? srt[midx] : 0.f;
  __syncthreads();

  float a1 = 0.f, a2 = 0.f;
  for (int e = tid; e < PATCH_; e += 256) {
    a1 += pm[e] * fabsf(pin[e] - med_in);
    a2 += pm[e] * fabsf(ptg[e] - med_tg);
  }
  float den = fmaxf((float)n, 1.f);
  float sc1 = fmaxf(block_sum(a1, red) / den, EPS_);
  float sc2 = fmaxf(block_sum(a2, red) / den, EPS_);

  float es = 0.f;
  for (int e = tid; e < PATCH_; e += 256) {
    float gi = (pin[e] - med_in) / sc1;
    float gt = (ptg[e] - med_tg) / sc2;
    es += pm[e] * fmaxf(fabsf(gi - gt), EPS_);
  }
  float ES = block_sum(es, red);
  if (tid == 0) {
    ep[b * K_ + p]     = sqrtf(fmaxf(ES / den, EPS_));
    pvalid[b * K_ + p] = (n >= 4) ? 1.f : 0.f;
  }
}

// ---------------- global median: 3-level radix select ----------------
__global__ void ghist_kernel(const float* __restrict__ inp, const float* __restrict__ tgt,
                             const unsigned char* __restrict__ msk,
                             unsigned* __restrict__ hist, const unsigned* __restrict__ pfx,
                             int level) {
  __shared__ unsigned h[4096];
  int b = blockIdx.y, tid = threadIdx.x;
  for (int t = tid; t < 4096; t += 256) h[t] = 0u;
  __syncthreads();
  unsigned p0 = 0u, p1 = 0u;
  if (level > 0) { p0 = pfx[b]; p1 = pfx[B_ + b]; }
  size_t base = (size_t)b * NTOT_ + (size_t)blockIdx.x * CHUNK_;
  for (int i = tid; i < CHUNK_; i += 256) {
    size_t g = base + i;
    if (msk[g]) {
      unsigned k0 = fkey(inp[g]);
      unsigned k1 = fkey(tgt[g]);
      bool ok0, ok1; unsigned b0, b1;
      if (level == 0)      { ok0 = true;                ok1 = true;
                             b0 = k0 >> 21;             b1 = k1 >> 21; }
      else if (level == 1) { ok0 = (k0 >> 21) == p0;    ok1 = (k1 >> 21) == p1;
                             b0 = (k0 >> 10) & 2047u;   b1 = (k1 >> 10) & 2047u; }
      else                 { ok0 = (k0 >> 10) == p0;    ok1 = (k1 >> 10) == p1;
                             b0 = k0 & 1023u;           b1 = k1 & 1023u; }
      if (ok0) atomicAdd(&h[b0], 1u);
      if (ok1) atomicAdd(&h[2048 + b1], 1u);
    }
  }
  __syncthreads();
  for (int t = tid; t < 4096; t += 256) {
    unsigned c = h[t];
    if (c) {
      int arr = t >> 11, bin = t & 2047;
      atomicAdd(&hist[((size_t)arr * B_ + b) * 2048 + bin], c);
    }
  }
}

__global__ void gscan_kernel(unsigned* __restrict__ hist, unsigned* __restrict__ pfx,
                             unsigned* __restrict__ kthr, unsigned* __restrict__ ncnt,
                             float* __restrict__ gmed, int level) {
  int b = blockIdx.x, tid = threadIdx.x;
  if (tid == 0) {
    for (int arr = 0; arr < 2; ++arr) {
      unsigned* h = hist + ((size_t)arr * B_ + b) * 2048;
      unsigned kth, n;
      if (level == 0) {
        unsigned tot = 0;
        for (int i = 0; i < 2048; ++i) tot += h[i];
        n = tot;
        if (arr == 0) ncnt[b] = tot;
        kth = (tot > 0) ? ((tot - 1) >> 1) : 0u;
      } else {
        kth = kthr[arr * B_ + b];
        n = ncnt[b];
      }
      unsigned sel = 0u, cum = 0u;
      if (n > 0) {
        for (int i = 0; i < 2048; ++i) {
          unsigned c = h[i];
          if (cum + c > kth) { sel = (unsigned)i; break; }
          cum += c;
        }
      }
      kthr[arr * B_ + b] = kth - cum;
      unsigned np = (level == 0) ? sel
                   : ((pfx[arr * B_ + b] << ((level == 2) ? 10 : 11)) | sel);
      pfx[arr * B_ + b] = np;
      if (level == 2) {
        unsigned keyv = np;                                  // full 32-bit key
        unsigned fb = (keyv & 0x80000000u) ? (keyv ^ 0x80000000u) : ~keyv;
        gmed[arr * B_ + b] = (n > 0) ? __uint_as_float(fb) : 0.f;
      }
    }
  }
  __syncthreads();
  for (int t = tid; t < 4096; t += blockDim.x) {             // re-zero for next level/call
    int arr = t >> 11, bin = t & 2047;
    hist[((size_t)arr * B_ + b) * 2048 + bin] = 0u;
  }
}

// ---------------- global scale & error reductions ----------------
__global__ void gscale_kernel(const float* __restrict__ inp, const float* __restrict__ tgt,
                              const unsigned char* __restrict__ msk,
                              const float* __restrict__ gmed, float* __restrict__ part) {
  __shared__ float red[12];
  int b = blockIdx.y, cx = blockIdx.x, tid = threadIdx.x;
  float m0 = gmed[b], m1 = gmed[B_ + b];
  size_t base = (size_t)b * NTOT_ + (size_t)cx * CHUNK_;
  float s1 = 0.f, s2 = 0.f;
  for (int i = tid; i < CHUNK_; i += 256) {
    size_t g = base + i;
    if (msk[g]) { s1 += fabsf(inp[g] - m0); s2 += fabsf(tgt[g] - m1); }
  }
  float S1 = block_sum(s1, red);
  float S2 = block_sum(s2, red);
  if (tid == 0) {
    part[(0 * B_ + b) * NCHUNK_ + cx] = S1;
    part[(1 * B_ + b) * NCHUNK_ + cx] = S2;
  }
}

__global__ void gscale_reduce_kernel(const float* __restrict__ part,
                                     const unsigned* __restrict__ ncnt,
                                     float* __restrict__ gsc) {
  __shared__ float red[12];
  int b = blockIdx.x, tid = threadIdx.x;
  float t1 = 0.f, t2 = 0.f;
  for (int i = tid; i < NCHUNK_; i += 256) {
    t1 += part[(0 * B_ + b) * NCHUNK_ + i];
    t2 += part[(1 * B_ + b) * NCHUNK_ + i];
  }
  float S1 = block_sum(t1, red);
  float S2 = block_sum(t2, red);
  if (tid == 0) {
    float den = fmaxf((float)ncnt[b], 1.f);
    gsc[b]      = fmaxf(S1 / den, EPS_);
    gsc[B_ + b] = fmaxf(S2 / den, EPS_);
  }
}

__global__ void gerr_kernel(const float* __restrict__ inp, const float* __restrict__ tgt,
                            const unsigned char* __restrict__ msk,
                            const float* __restrict__ gmed, const float* __restrict__ gsc,
                            float* __restrict__ part) {
  __shared__ float red[12];
  int b = blockIdx.y, cx = blockIdx.x, tid = threadIdx.x;
  float m0 = gmed[b], m1 = gmed[B_ + b];
  float c0 = gsc[b],  c1 = gsc[B_ + b];
  size_t base = (size_t)b * NTOT_ + (size_t)cx * CHUNK_;
  float es = 0.f;
  for (int i = tid; i < CHUNK_; i += 256) {
    size_t g = base + i;
    if (msk[g]) {
      float gi = (inp[g] - m0) / c0;
      float gt = (tgt[g] - m1) / c1;
      es += fmaxf(fabsf(gi - gt), EPS_);
    }
  }
  float ES = block_sum(es, red);
  if (tid == 0) part[(2 * B_ + b) * NCHUNK_ + cx] = ES;
}

__global__ void gfinal_kernel(const float* __restrict__ part, const unsigned* __restrict__ ncnt,
                              const float* __restrict__ ep, const float* __restrict__ pvalid,
                              float* __restrict__ out) {
  __shared__ float red[12];
  int b = blockIdx.x, tid = threadIdx.x;
  float t = 0.f;
  for (int i = tid; i < NCHUNK_; i += 256) t += part[(2 * B_ + b) * NCHUNK_ + i];
  float ES = block_sum(t, red);
  float se = 0.f, sv = 0.f;
  for (int i = tid; i < K_; i += 256) {
    float v = pvalid[b * K_ + i];
    se += ep[b * K_ + i] * v;
    sv += v;
  }
  float SE = block_sum(se, red);
  float SV = block_sum(sv, red);
  if (tid == 0) {
    float den = fmaxf((float)ncnt[b], 1.f);
    float e_glob  = sqrtf(fmaxf(ES / den, EPS_));
    float e_patch = SE / fmaxf(SV, 1.f);
    out[b] = 0.5f * (e_patch + e_glob);
  }
}

// ---------------- host-side launch sequence ----------------
extern "C" void kernel_launch(void* const* d_in, const int* in_sizes, int n_in,
                              void* d_out, int out_size, void* d_ws, size_t ws_size,
                              hipStream_t stream) {
  (void)in_sizes; (void)n_in; (void)out_size; (void)ws_size;
  const float*         inp = (const float*)d_in[0];
  const float*         tgt = (const float*)d_in[1];
  const unsigned char* msk = (const unsigned char*)d_in[2];
  const float*         img = (const float*)d_in[3];
  const unsigned char* vmk = (const unsigned char*)d_in[4];
  float* out = (float*)d_out;

  // workspace layout (all 4-byte types)
  float*    e_rs   = (float*)d_ws;                      // B*NP
  int*      coords = (int*)(e_rs + B_ * NP_);           // B*K*2
  float*    ep     = (float*)(coords + B_ * K_ * 2);    // B*K
  float*    pval   = ep + B_ * K_;                      // B*K
  unsigned* hist   = (unsigned*)(pval + B_ * K_);       // 2*B*2048
  unsigned* pfx    = hist + 2 * B_ * 2048;              // 2*B
  unsigned* kthr   = pfx + 2 * B_;                      // 2*B
  unsigned* ncnt   = kthr + 2 * B_;                     // B
  float*    gmed   = (float*)(ncnt + B_);               // 2*B
  float*    gsc    = gmed + 2 * B_;                     // 2*B
  float*    part   = gsc + 2 * B_;                      // 3*B*NCHUNK

  const int histN = 2 * B_ * 2048;
  zero_hist_kernel<<<(histN + 255) / 256, 256, 0, stream>>>(hist, histN);

  resize_edges_kernel<<<dim3((NP_ + 255) / 256, B_), 256, 0, stream>>>(img, vmk, e_rs);
  topk_kernel<<<B_, 1024, 0, stream>>>(e_rs, coords);
  patch_kernel<<<dim3(K_, B_), 256, 0, stream>>>(inp, tgt, msk, coords, ep, pval);

  for (int level = 0; level < 3; ++level) {
    ghist_kernel<<<dim3(NCHUNK_, B_), 256, 0, stream>>>(inp, tgt, msk, hist, pfx, level);
    gscan_kernel<<<B_, 256, 0, stream>>>(hist, pfx, kthr, ncnt, gmed, level);
  }

  gscale_kernel<<<dim3(NCHUNK_, B_), 256, 0, stream>>>(inp, tgt, msk, gmed, part);
  gscale_reduce_kernel<<<B_, 256, 0, stream>>>(part, ncnt, gsc);
  gerr_kernel<<<dim3(NCHUNK_, B_), 256, 0, stream>>>(inp, tgt, msk, gmed, gsc, part);
  gfinal_kernel<<<B_, 256, 0, stream>>>(part, ncnt, ep, pval, out);
}
